// CrossEntropyPAUCLoss_79130477461783
// MI455X (gfx1250) — compile-verified, matrix operand order checked by reading, not run on hardware
//
#include <hip/hip_runtime.h>
#include <stdint.h>

// ---------------------------------------------------------------------------
// CrossEntropy + macro pAUC loss for MI455X (gfx1250, wave32).
//
//   1) ce_keys: f32 softmax stats exactly; WMMA bf16 ones-matmul for the
//      0.1-weighted smoothing sum; emits per-class u32 sort keys (order-
//      preserving float transform, positive-flag in bit 0), class-major.
//   2) bitonic sort (ascending u32 == descending log-prob):
//        local superpass (128 KB LDS tile, stages k=2..32768),
//        global j=32768, local k=65536, global j=65536,32768, local k=131072.
//      Tile staging uses GLOBAL_LOAD_ASYNC_TO_LDS_B128 / STORE_ASYNC via the
//      gfx1250 builtins (ASYNCcnt DMA), falling back to sync b128 copies.
//   3) pauc_scan: one 16-wave block per class, two passes:
//        pass1: per-segment positive counts (also yields P), LDS prefix
//        pass2: ballot/popcount scan; pauc_c=(sum_{j<=J} t_j + frac*t_{J+1})
//               /(P*Nneg), J=floor(0.7*Nneg)
//   4) finalize: out = 0.5*ce_mean + 0.5*(1 - pauc^2)
//
// ws layout: [0..16) floats: 0=ce_sum 1=pauc_sum 2=valid_cnt
//            byte offset 4096: u32 items[128][131072] (64 MB)
// ---------------------------------------------------------------------------

#define NS 131072
#define NC 128
#define TILE_SORT 32768
#define SORT_THREADS 1024
#define ITEMS_OFF 4096
#define PAUC_WAVES 16
#define PAUC_THREADS (PAUC_WAVES * 32)
#define SEG (NS / PAUC_WAVES)

typedef __attribute__((ext_vector_type(16))) __bf16 v16bf;
typedef __attribute__((ext_vector_type(8)))  float  v8f;
typedef __attribute__((ext_vector_type(4)))  int    v4i;

#define AS1 __attribute__((address_space(1)))
#define AS3 __attribute__((address_space(3)))

#if __has_builtin(__builtin_amdgcn_global_load_async_to_lds_b128) && \
    __has_builtin(__builtin_amdgcn_global_store_async_from_lds_b128)
#define HAS_ASYNC_LDS 1
#else
#define HAS_ASYNC_LDS 0
#endif

__device__ inline void wait_async0() {
#if HAS_ASYNC_LDS
#if __has_builtin(__builtin_amdgcn_s_wait_asynccnt)
  __builtin_amdgcn_s_wait_asynccnt(0);
#else
  asm volatile("s_wait_asynccnt 0" ::: "memory");
#endif
#endif
}

// 16-byte global->LDS copy (ASYNCcnt DMA on gfx1250).
__device__ inline void copy16_g2l(uint32_t* lds, const uint32_t* g) {
#if HAS_ASYNC_LDS
  __builtin_amdgcn_global_load_async_to_lds_b128((AS1 v4i*)g, (AS3 v4i*)lds, 0, 0);
#else
  *(uint4*)lds = *(const uint4*)g;
#endif
}
// 16-byte LDS->global copy (ASYNCcnt DMA on gfx1250).
__device__ inline void copy16_l2g(uint32_t* g, const uint32_t* lds) {
#if HAS_ASYNC_LDS
  __builtin_amdgcn_global_store_async_from_lds_b128((AS1 v4i*)g, (AS3 v4i*)lds, 0, 0);
#else
  *(uint4*)g = *(const uint4*)lds;
#endif
}

__global__ __launch_bounds__(256) void init_accum_kernel(float* __restrict__ accum) {
  int i = threadIdx.x;
  if (i < 16) accum[i] = 0.0f;
}

// One wave (32 lanes) per 16-row tile.
__global__ __launch_bounds__(32) void ce_keys_kernel(const float* __restrict__ pred,
                                                     const int* __restrict__ tgt,
                                                     uint32_t* __restrict__ items,
                                                     float* __restrict__ ce_sum) {
  __shared__ __align__(16) float xs[16 * NC];
  __shared__ float mxs[16];
  __shared__ float lzs[16];
  __shared__ int   tg[16];

  const int lane    = threadIdx.x;
  const int rowBase = blockIdx.x * 16;
  const int m       = lane & 15;
  const int half    = lane >> 4;

  // stage 16x128 f32 tile into LDS (async DMA path on gfx1250)
  const float* src = pred + (size_t)rowBase * NC;
  for (int t = lane * 4; t < 16 * NC; t += 32 * 4)
    copy16_g2l((uint32_t*)&xs[t], (const uint32_t*)&src[t]);
  wait_async0();
  __syncthreads();

  // exact f32 row max and sum-exp: lanes (m, m+16) split the 128 columns
  float mx = -3.4e38f;
  for (int c = half * 64; c < half * 64 + 64; ++c) mx = fmaxf(mx, xs[m * NC + c]);
  mx = fmaxf(mx, __shfl_xor(mx, 16, 32));
  float s1 = 0.0f;
  for (int c = half * 64; c < half * 64 + 64; ++c) s1 += __expf(xs[m * NC + c] - mx);
  s1 += __shfl_xor(s1, 16, 32);
  float lz = __logf(s1);
  if (half == 0) { mxs[m] = mx; lzs[m] = lz; tg[m] = tgt[rowBase + m]; }
  __syncthreads();

  // WMMA: S2[m] = sum_c (x[m,c]-mx[m])  (0.1-weighted smoothing term, bf16 OK)
  v16bf ones;
  for (int j = 0; j < 16; ++j) ones[j] = (__bf16)1.0f;
  v8f d;
  for (int r = 0; r < 8; ++r) d[r] = 0.0f;
  for (int t = 0; t < 4; ++t) {
    v16bf a;
    for (int j = 0; j < 16; ++j) {
      int K = j + 8 * half + ((j >= 8) ? 8 : 0);
      a[j] = (__bf16)(xs[m * NC + (32 * t + K)] - mx);
    }
    d = __builtin_amdgcn_wmma_f32_16x16x32_bf16(false, a, false, ones,
                                                (short)0, d, false, false);
  }

  // D layout: lane 0 holds rows 0-7 in d[0..7], lane 16 rows 8-15.
  if ((lane & 15) == 0) {
    float part = 0.0f;
    for (int r = 0; r < 8; ++r) {
      int   mm     = r + 8 * half;
      float S2     = d[r];
      float nll    = -(xs[mm * NC + tg[mm]] - mxs[mm]) + lzs[mm];
      float smooth = -S2 * (1.0f / NC) + lzs[mm];
      part += 0.9f * nll + 0.1f * smooth;
    }
    atomicAdd(ce_sum, part);
  }
  __syncthreads();

  // Emit sort keys: ascending u32 == descending log-prob; flag in bit 0.
  const int tgm = tg[m];
  for (int p = 0; p < 64; ++p) {
    int      c    = 2 * p + half;
    float    logp = xs[m * NC + c] - mx - lz;
    uint32_t u    = __float_as_uint(logp);
    uint32_t mono = u ^ ((u >> 31) ? 0xFFFFFFFFu : 0x80000000u);
    uint32_t key  = ~mono;
    uint32_t flag = (tgm == c) ? 1u : 0u;
    items[(size_t)c * NS + rowBase + m] = (key & ~1u) | flag;
  }
}

// Local bitonic phases on a 32768-element (128 KB) LDS tile.
__global__ __launch_bounds__(SORT_THREADS) void sort_local_kernel(uint32_t* __restrict__ items,
                                                                  uint32_t kLo, uint32_t kHi) {
  __shared__ __align__(16) uint32_t sm[TILE_SORT];
  const int tilesPerClass = NS / TILE_SORT;
  const int cls  = blockIdx.x / tilesPerClass;
  const int tile = blockIdx.x % tilesPerClass;
  uint32_t* base = items + (size_t)cls * NS + (size_t)tile * TILE_SORT;
  const uint32_t gBase = (uint32_t)tile * TILE_SORT;

  for (int t = threadIdx.x * 4; t < TILE_SORT; t += SORT_THREADS * 4)
    copy16_g2l(&sm[t], &base[t]);
  wait_async0();
  __syncthreads();

  for (uint32_t k = kLo; k <= kHi && k != 0; k <<= 1) {
    uint32_t jStart = (k >> 1) > (TILE_SORT >> 1) ? (TILE_SORT >> 1) : (k >> 1);
    for (uint32_t j = jStart; j > 0; j >>= 1) {
      if (j >= 4) {
        // 4 contiguous pairs per thread: b128 LDS traffic, uniform direction
        for (uint32_t p = threadIdx.x * 4u; p < (TILE_SORT / 2); p += SORT_THREADS * 4u) {
          uint32_t i = 2u * (p & ~(j - 1u)) + (p & (j - 1u));
          bool asc = (((gBase + i) & k) == 0u);
          uint4 a = *(uint4*)&sm[i];
          uint4 b = *(uint4*)&sm[i + j];
          uint32_t t;
          if ((a.x > b.x) == asc) { t = a.x; a.x = b.x; b.x = t; }
          if ((a.y > b.y) == asc) { t = a.y; a.y = b.y; b.y = t; }
          if ((a.z > b.z) == asc) { t = a.z; a.z = b.z; b.z = t; }
          if ((a.w > b.w) == asc) { t = a.w; a.w = b.w; b.w = t; }
          *(uint4*)&sm[i] = a;
          *(uint4*)&sm[i + j] = b;
        }
      } else {
        for (uint32_t p = threadIdx.x; p < (TILE_SORT / 2); p += SORT_THREADS) {
          uint32_t i = 2u * (p & ~(j - 1u)) + (p & (j - 1u));
          uint32_t l = i + j;
          bool asc = (((gBase + i) & k) == 0u);
          uint32_t a = sm[i], b = sm[l];
          if ((a > b) == asc) { sm[i] = b; sm[l] = a; }
        }
      }
      __syncthreads();
    }
  }
  for (int t = threadIdx.x * 4; t < TILE_SORT; t += SORT_THREADS * 4)
    copy16_l2g(&base[t], &sm[t]);
  wait_async0();
}

// One global pass (stride j >= tile size): 4 contiguous pairs per thread.
__global__ __launch_bounds__(256) void sort_global_kernel(uint32_t* __restrict__ items,
                                                          uint32_t k, uint32_t j) {
  uint32_t tid = blockIdx.x * 256u + threadIdx.x;
  uint32_t pp  = tid * 4u;                 // first of 4 pair indices
  uint32_t cls = pp >> 16;                 // NS/2 == 2^16 pairs per class
  uint32_t p   = pp & (NS / 2 - 1u);
  uint32_t i = 2u * (p & ~(j - 1u)) + (p & (j - 1u));
  uint32_t l = i + j;
  bool asc = ((i & k) == 0u);              // uniform across the 4 (j >= 4)
  uint32_t* b = items + (size_t)cls * NS;
  uint4 a = *(uint4*)&b[i];
  uint4 c = *(uint4*)&b[l];
  uint32_t t;
  if ((a.x > c.x) == asc) { t = a.x; a.x = c.x; c.x = t; }
  if ((a.y > c.y) == asc) { t = a.y; a.y = c.y; c.y = t; }
  if ((a.z > c.z) == asc) { t = a.z; a.z = c.z; c.z = t; }
  if ((a.w > c.w) == asc) { t = a.w; a.w = c.w; c.w = t; }
  *(uint4*)&b[i] = a;
  *(uint4*)&b[l] = c;
}

// One 16-wave block per class; two passes over the sorted keys.
__global__ __launch_bounds__(PAUC_THREADS) void pauc_scan_kernel(const uint32_t* __restrict__ items,
                                                                 float* __restrict__ pauc_sum,
                                                                 float* __restrict__ valid_sum) {
  __shared__ uint32_t segPos[PAUC_WAVES];
  __shared__ unsigned long long accW[PAUC_WAVES];
  __shared__ uint32_t tbW[PAUC_WAVES];

  const int cls  = blockIdx.x;
  const int lane = threadIdx.x & 31;
  const int wv   = threadIdx.x >> 5;
  const uint32_t* seg = items + (size_t)cls * NS + (size_t)wv * SEG;
  const uint32_t lt = (1u << lane) - 1u;

  // pass 1: positives per segment (uint4 per lane -> 128 elems per wave-iter)
  uint32_t pos = 0;
#pragma unroll 4
  for (int it = 0; it < SEG; it += 128) {
    uint4 v = *(const uint4*)&seg[it + lane * 4];
    pos += (v.x & 1u) + (v.y & 1u) + (v.z & 1u) + (v.w & 1u);
  }
  for (int o = 16; o > 0; o >>= 1) pos += __shfl_xor(pos, o, 32);
  if (lane == 0) segPos[wv] = pos;
  __syncthreads();

  uint32_t P = 0, tp0 = 0;
  for (int w = 0; w < PAUC_WAVES; ++w) { P += segPos[w]; if (w < wv) tp0 += segPos[w]; }
  const uint32_t Nneg = NS - P;
  const uint32_t J    = (uint32_t)(0.7 * (double)Nneg);
  const double   frac = 0.7 * (double)Nneg - (double)J;
  const uint32_t fp0  = (uint32_t)wv * SEG - tp0;

  // pass 2: contribution scan with exact per-wave offsets
  uint32_t tp = tp0, fp = fp0, tb = 0;
  unsigned long long acc = 0ull;
#pragma unroll 2
  for (int it = 0; it < SEG; it += 128) {
    __builtin_prefetch(seg + it + lane * 4 + 2048, 0, 0);
    uint4 v = *(const uint4*)&seg[it + lane * 4];
    uint32_t f0 = v.x & 1u, f1 = v.y & 1u, f2 = v.z & 1u, f3 = v.w & 1u;
    uint32_t m0 = __builtin_amdgcn_ballot_w32(f0 != 0u);
    uint32_t m1 = __builtin_amdgcn_ballot_w32(f1 != 0u);
    uint32_t m2 = __builtin_amdgcn_ballot_w32(f2 != 0u);
    uint32_t m3 = __builtin_amdgcn_ballot_w32(f3 != 0u);
    uint32_t below = __popc(m0 & lt) + __popc(m1 & lt) + __popc(m2 & lt) + __popc(m3 & lt);
    uint32_t tpq = tp + below;                 // before this lane's component 0
    uint32_t fpq = fp + (4u * (uint32_t)lane - below);
    uint32_t fl[4] = { f0, f1, f2, f3 };
#pragma unroll
    for (int w = 0; w < 4; ++w) {
      if (!fl[w]) {
        uint32_t jn = fpq + 1u;
        if (jn <= J) acc += (unsigned long long)tpq;
        else if (jn == J + 1u) tb = tpq;       // crossing-segment TP count
        fpq++;
      } else {
        tpq++;
      }
    }
    uint32_t cpos = __popc(m0) + __popc(m1) + __popc(m2) + __popc(m3);
    tp += cpos;
    fp += 128u - cpos;
  }
  for (int o = 16; o > 0; o >>= 1) {
    acc += __shfl_down(acc, o, 32);
    tb  += __shfl_down(tb, o, 32);             // at most one nonzero lane overall
  }
  if (lane == 0) { accW[wv] = acc; tbW[wv] = tb; }
  __syncthreads();

  if (threadIdx.x == 0 && P > 0u) {
    unsigned long long A = 0ull;
    uint32_t T = 0;
    for (int w = 0; w < PAUC_WAVES; ++w) { A += accW[w]; T += tbW[w]; }
    double pc = ((double)A + frac * (double)T) / ((double)P * (double)Nneg);
    atomicAdd(pauc_sum, (float)pc);
    atomicAdd(valid_sum, 1.0f);
  }
}

__global__ void finalize_kernel(const float* __restrict__ accum, float* __restrict__ out) {
  float ce   = accum[0] * (1.0f / (float)NS);
  float v    = fmaxf(accum[2], 1.0f);
  float pauc = accum[1] / v;
  out[0] = 0.5f * ce + 0.5f * (1.0f - pauc * pauc);
}

extern "C" void kernel_launch(void* const* d_in, const int* in_sizes, int n_in,
                              void* d_out, int out_size, void* d_ws, size_t ws_size,
                              hipStream_t stream) {
  const float* pred = (const float*)d_in[0];
  const int*   tgt  = (const int*)d_in[1];
  float*       out  = (float*)d_out;

  float*    accum = (float*)d_ws;                         // [0]=ce [1]=pauc [2]=valid
  uint32_t* items = (uint32_t*)((char*)d_ws + ITEMS_OFF); // 64 MB sort keys

  init_accum_kernel<<<1, 256, 0, stream>>>(accum);
  ce_keys_kernel<<<NS / 16, 32, 0, stream>>>(pred, tgt, items, accum);

  const int gridL = NC * (NS / TILE_SORT);                // 512
  const int gridG = (NC * (NS / 2)) / (256 * 4);          // 8192

  sort_local_kernel<<<gridL, SORT_THREADS, 0, stream>>>(items, 2u, (uint32_t)TILE_SORT);
  sort_global_kernel<<<gridG, 256, 0, stream>>>(items, 65536u, 32768u);
  sort_local_kernel<<<gridL, SORT_THREADS, 0, stream>>>(items, 65536u, 65536u);
  sort_global_kernel<<<gridG, 256, 0, stream>>>(items, 131072u, 65536u);
  sort_global_kernel<<<gridG, 256, 0, stream>>>(items, 131072u, 32768u);
  sort_local_kernel<<<gridL, SORT_THREADS, 0, stream>>>(items, 131072u, 131072u);

  pauc_scan_kernel<<<NC, PAUC_THREADS, 0, stream>>>(items, accum + 1, accum + 2);
  finalize_kernel<<<1, 1, 0, stream>>>(accum, out);
}
